// PropNetDiffDenModel_57861799411924
// MI455X (gfx1250) — matrix operand order repaired and checked by reading, not compiled, not run on hardware
//
#include <hip/hip_runtime.h>
#include <hip/hip_bf16.h>
#include <stdint.h>

// ---------------------------------------------------------------------------
// PropNet forward for MI455X (gfx1250, wave32).
// All dense layers run on v_wmma_f32_16x16x32_bf16 (fp32 accumulate).
// Weights are pre-swizzled into per-lane WMMA fragment layout so every B
// operand is one contiguous 32B v16bf load; A operands are b128 float4 loads.
// The relation-propagation GEMM gathers [rel_enc | v*eff_r | v*eff_s] on the
// fly (A fragments are per-lane rows), eliminating the 192-wide concat buffer.
// ---------------------------------------------------------------------------

typedef __attribute__((ext_vector_type(16))) __bf16 v16bf;
typedef __attribute__((ext_vector_type(8)))  float  v8f;

static constexpr int B_   = 2;
static constexpr int N_   = 2048;
static constexpr int TOPK = 8;
static constexpr int R_   = N_ * TOPK;      // 16384 relations per batch
static constexpr int NF   = 64;
static constexpr float T2 = 0.12f * 0.12f;  // ADJ_THRESH^2

__device__ __forceinline__ v8f wmma_bf16(v16bf a, v16bf b, v8f c) {
  return __builtin_amdgcn_wmma_f32_16x16x32_bf16(false, a, false, b, (short)0, c,
                                                 false, false);
}

// pack 16 fp32 (scaled) into a bf16 A-fragment; base points at k0+8*half
__device__ __forceinline__ v16bf pack_a(const float* base, float sc) {
  const float4* ap = (const float4*)base;
  float4 f0 = ap[0], f1 = ap[1];   // K +0..7
  float4 f2 = ap[4], f3 = ap[5];   // K +16..23
  v16bf a;
  a[0]  = (__bf16)(sc * f0.x); a[1]  = (__bf16)(sc * f0.y);
  a[2]  = (__bf16)(sc * f0.z); a[3]  = (__bf16)(sc * f0.w);
  a[4]  = (__bf16)(sc * f1.x); a[5]  = (__bf16)(sc * f1.y);
  a[6]  = (__bf16)(sc * f1.z); a[7]  = (__bf16)(sc * f1.w);
  a[8]  = (__bf16)(sc * f2.x); a[9]  = (__bf16)(sc * f2.y);
  a[10] = (__bf16)(sc * f2.z); a[11] = (__bf16)(sc * f2.w);
  a[12] = (__bf16)(sc * f3.x); a[13] = (__bf16)(sc * f3.y);
  a[14] = (__bf16)(sc * f3.z); a[15] = (__bf16)(sc * f3.w);
  return a;
}

// --- weight pre-swizzle: fp32 [K0 x N0] -> bf16 fragment-major ---------------
// frag[(((kc*NT + t)*32) + lane)*16 + e] = W[kc*32 + 8*(lane>>4) + (e&7) + 16*(e>>3)][t*16 + (lane&15)]
__global__ void swz_weight_kernel(const float* __restrict__ src, __bf16* __restrict__ dst,
                                  int K0, int N0, int KC, int NT) {
  int x = blockIdx.x * blockDim.x + threadIdx.x;
  if (x >= KC * NT * 512) return;
  int e    = x & 15;
  int lane = (x >> 4) & 31;
  int rest = x >> 9;
  int t  = rest % NT;
  int kc = rest / NT;
  int k = kc * 32 + 8 * (lane >> 4) + (e & 7) + 16 * (e >> 3);
  int n = t * 16 + (lane & 15);
  float v = (k < K0 && n < N0) ? src[k * N0 + n] : 0.0f;
  dst[x] = (__bf16)v;
}

// --- edge construction: per-receiver top-8 over LDS-staged states ------------
__global__ void topk_kernel(const float* __restrict__ states,
                            const uint8_t* __restrict__ mask,
                            const uint8_t* __restrict__ tool,
                            int* __restrict__ idx, float* __restrict__ valid) {
  __shared__ float  sx[N_], sy[N_], sz[N_];
  __shared__ uint8_t sm[N_], st[N_];
  const int PER = 256;
  int b = blockIdx.x / (N_ / PER);
  int i = (blockIdx.x % (N_ / PER)) * PER + threadIdx.x;
  const float* S = states + (size_t)b * N_ * 3;
  for (int j = threadIdx.x; j < N_; j += PER) {
    sx[j] = S[j * 3 + 0]; sy[j] = S[j * 3 + 1]; sz[j] = S[j * 3 + 2];
    sm[j] = mask[b * N_ + j]; st[j] = tool[b * N_ + j];
  }
  __syncthreads();
  float px = sx[i], py = sy[i], pz = sz[i];
  bool mi = sm[i] != 0, ti = st[i] != 0;
  float bd[TOPK]; int bi[TOPK];
#pragma unroll
  for (int k = 0; k < TOPK; ++k) { bd[k] = 3.0e38f; bi[k] = 0; }
  for (int j = 0; j < N_; ++j) {
    float dx = px - sx[j], dy = py - sy[j], dz = pz - sz[j];
    float d = dx * dx + dy * dy + dz * dz;
    bool ok = mi && (sm[j] != 0) && !(ti && (st[j] != 0));  // tool-tool pairs cut
    d = ok ? d : 1.0e10f;
    if (d < bd[TOPK - 1]) {             // strict <: lower index wins ties (lax.top_k)
      int p = TOPK - 1;
#pragma unroll
      for (int q = TOPK - 1; q > 0; --q)
        if (d < bd[q - 1]) { bd[q] = bd[q - 1]; bi[q] = bi[q - 1]; p = q - 1; }
      bd[p] = d; bi[p] = j;
    }
  }
  int*   ip = idx   + ((size_t)b * N_ + i) * TOPK;
  float* vp = valid + ((size_t)b * N_ + i) * TOPK;
#pragma unroll
  for (int k = 0; k < TOPK; ++k) {
    ip[k] = bi[k];
    vp[k] = (mi && !ti && bd[k] < T2) ? 1.0f : 0.0f;  // tool receivers: no edges
  }
}

// --- particle-encoder input: [s_delta (9) | a_cur x3 | zero-pad] to K=32 -----
__global__ void build_x_kernel(const float* __restrict__ s_delta,
                               const float* __restrict__ a_cur,
                               float* __restrict__ X32) {
  int m = blockIdx.x * blockDim.x + threadIdx.x;
  if (m >= B_ * N_) return;
  int b = m / N_, n = m % N_;
  float* row = X32 + (size_t)m * 32;
#pragma unroll
  for (int c = 0; c < 9; ++c) row[c] = s_delta[((size_t)b * 9 + c) * N_ + n];
  float a = a_cur[m];
  row[9] = a; row[10] = a; row[11] = a;
#pragma unroll
  for (int c = 12; c < 32; ++c) row[c] = 0.0f;
}

// --- relation-encoder input: [a_r, a_s, s_r - s_s | zero-pad] to K=32 --------
__global__ void build_relin_kernel(const float* __restrict__ states,
                                   const float* __restrict__ a_cur,
                                   const int* __restrict__ idx,
                                   const float* __restrict__ valid,
                                   float* __restrict__ RI) {
  int m = blockIdx.x * blockDim.x + threadIdx.x;
  if (m >= B_ * R_) return;
  int b = m / R_, r = m % R_;
  int i = r / TOPK;
  float v = valid[m];
  int j = idx[m];
  const float* S = states + (size_t)b * N_ * 3;
  float* row = RI + (size_t)m * 32;
  row[0] = v * a_cur[b * N_ + i];
  row[1] = v * a_cur[b * N_ + j];
  row[2] = v * (S[i * 3 + 0] - S[j * 3 + 0]);
  row[3] = v * (S[i * 3 + 1] - S[j * 3 + 1]);
  row[4] = v * (S[i * 3 + 2] - S[j * 3 + 2]);
#pragma unroll
  for (int c = 5; c < 32; ++c) row[c] = 0.0f;
}

// --- agg = Rr^T @ rel_eff : contiguous 8-slot sum per particle ---------------
__global__ void aggregate_kernel(const float* __restrict__ releff,
                                 const float* __restrict__ valid,
                                 float* __restrict__ ppcat) {
  int m = blockIdx.x * blockDim.x + threadIdx.x;
  if (m >= B_ * N_) return;
  int b = m / N_, i = m % N_;
  const float* vv = valid + (size_t)m * TOPK;
  float* dst = ppcat + (size_t)m * 128 + 64;
  for (int c = 0; c < NF; c += 4) {
    float4 acc = make_float4(0.f, 0.f, 0.f, 0.f);
#pragma unroll
    for (int k = 0; k < TOPK; ++k) {
      const float4 e =
          *(const float4*)(releff + ((size_t)b * R_ + (size_t)i * TOPK + k) * NF + c);
      float v = vv[k];
      acc.x += v * e.x; acc.y += v * e.y; acc.z += v * e.z; acc.w += v * e.w;
    }
    *(float4*)(dst + c) = acc;
  }
}

__global__ void copy_penc_kernel(const float* __restrict__ effect,
                                 float* __restrict__ ppcat) {
  int t = blockIdx.x * blockDim.x + threadIdx.x;
  if (t >= B_ * N_ * NF) return;
  int m = t / NF, c = t % NF;
  ppcat[(size_t)m * 128 + c] = effect[t];
}

// --- generic bf16-WMMA GEMM: one wave = 32 x (NT*16) stripe ------------------
// C = [ReLU]( A(MxK fp32) @ W(swizzled bf16) + bias [+ resid] )
template <int NT, bool RELU, bool RESID>
__global__ void gemm_kernel(const float* __restrict__ A, int lda,
                            const __bf16* __restrict__ Wf,   // fragment-major
                            const float* __restrict__ bias,  // NT*16
                            float* __restrict__ C, int ldc,
                            const float* __restrict__ resid, int ldr,
                            int M, int K) {
  int wave = (blockIdx.x * blockDim.x + threadIdx.x) >> 5;
  int lane = threadIdx.x & 31;
  int m0 = wave * 32;
  if (m0 >= M) return;
  int half = lane >> 4;
  int l16  = lane & 15;
  v8f acc[2][NT];
#pragma unroll
  for (int s = 0; s < 2; ++s)
#pragma unroll
    for (int t = 0; t < NT; ++t)
#pragma unroll
      for (int e = 0; e < 8; ++e) acc[s][t][e] = 0.0f;

  const float* arow0 = A + (size_t)(m0 + l16) * lda + 8 * half;
  const float* arow1 = arow0 + (size_t)16 * lda;
  for (int k0 = 0; k0 < K; k0 += 32) {
    if (k0 + 32 < K) {  // prefetch next A chunk (global_prefetch_b8)
      __builtin_prefetch((const void*)(arow0 + k0 + 32), 0, 1);
      __builtin_prefetch((const void*)(arow1 + k0 + 32), 0, 1);
    }
    v16bf af0 = pack_a(arow0 + k0, 1.0f);
    v16bf af1 = pack_a(arow1 + k0, 1.0f);
    int kc = k0 >> 5;
#pragma unroll
    for (int t = 0; t < NT; ++t) {
      v16bf bf_ = *(const v16bf*)(Wf + ((size_t)(kc * NT + t) * 32 + lane) * 16);
      acc[0][t] = wmma_bf16(af0, bf_, acc[0][t]);
      acc[1][t] = wmma_bf16(af1, bf_, acc[1][t]);
    }
  }
#pragma unroll
  for (int s = 0; s < 2; ++s)
#pragma unroll
    for (int t = 0; t < NT; ++t) {
      int n = t * 16 + l16;
      float bv = bias[n];
#pragma unroll
      for (int rr = 0; rr < 8; ++rr) {
        int row = m0 + s * 16 + rr + 8 * half;
        float v = acc[s][t][rr] + bv;
        if (RESID) v += resid[(size_t)row * ldr + n];
        if (RELU)  v = fmaxf(v, 0.0f);
        C[(size_t)row * ldc + n] = v;
      }
    }
}

// --- fused relation-propagation GEMM (K = 192, fully unrolled) ---------------
// A row r = [ rel_enc[r] | valid[r]*effect[recv(r)] | valid[r]*effect[send(r)] ]
__global__ void rel_gemm_kernel(const float* __restrict__ relenc,  // BR x 64
                                const float* __restrict__ effect,  // BN x 64
                                const int* __restrict__ idx,
                                const float* __restrict__ valid,
                                const __bf16* __restrict__ Wf,     // 192x64 swizzled
                                const float* __restrict__ bias,
                                float* __restrict__ C) {           // BR x 64
  int wave = (blockIdx.x * blockDim.x + threadIdx.x) >> 5;
  int lane = threadIdx.x & 31;
  int m0 = wave * 32;
  if (m0 >= B_ * R_) return;
  int half = lane >> 4;
  int l16  = lane & 15;

  const float* p0[2]; const float* p1[2]; const float* p2[2]; float vs[2];
#pragma unroll
  for (int s = 0; s < 2; ++s) {
    int row = m0 + s * 16 + l16;
    int b = row / R_, r = row % R_;
    int i = r / TOPK;
    vs[s] = valid[row];
    int j = idx[row];
    p0[s] = relenc + (size_t)row * NF + 8 * half;
    p1[s] = effect + ((size_t)b * N_ + i) * NF + 8 * half;
    p2[s] = effect + ((size_t)b * N_ + j) * NF + 8 * half;
  }

  v8f acc[2][4];
#pragma unroll
  for (int s = 0; s < 2; ++s)
#pragma unroll
    for (int t = 0; t < 4; ++t)
#pragma unroll
      for (int e = 0; e < 8; ++e) acc[s][t][e] = 0.0f;

#pragma unroll
  for (int kc = 0; kc < 6; ++kc) {        // 6 chunks of 32 over K=192
    const int seg = kc >> 1;              // 0: rel_enc, 1: eff_r, 2: eff_s
    const int off = (kc & 1) * 32;
    v16bf af[2];
#pragma unroll
    for (int s = 0; s < 2; ++s) {
      const float* base = (seg == 0) ? p0[s] : (seg == 1) ? p1[s] : p2[s];
      float sc = (seg == 0) ? 1.0f : vs[s];
      af[s] = pack_a(base + off, sc);
    }
#pragma unroll
    for (int t = 0; t < 4; ++t) {
      v16bf bf_ = *(const v16bf*)(Wf + ((size_t)(kc * 4 + t) * 32 + lane) * 16);
      acc[0][t] = wmma_bf16(af[0], bf_, acc[0][t]);
      acc[1][t] = wmma_bf16(af[1], bf_, acc[1][t]);
    }
  }
#pragma unroll
  for (int s = 0; s < 2; ++s)
#pragma unroll
    for (int t = 0; t < 4; ++t) {
      int n = t * 16 + l16;
      float bv = bias[n];
#pragma unroll
      for (int rr = 0; rr < 8; ++rr) {
        int row = m0 + s * 16 + rr + 8 * half;
        C[(size_t)row * NF + n] = fmaxf(acc[s][t][rr] + bv, 0.0f);
      }
    }
}

// --- final layer: out = states + H @ pr_w2 + pr_b2  (N padded 3 -> 16) -------
__global__ void final_kernel(const float* __restrict__ H,      // BN x 64
                             const __bf16* __restrict__ W2f,   // 64x16 swizzled
                             const float* __restrict__ b2,     // 3
                             const float* __restrict__ states,
                             float* __restrict__ out) {
  int wave = (blockIdx.x * blockDim.x + threadIdx.x) >> 5;
  int lane = threadIdx.x & 31;
  int m0 = wave * 16;
  if (m0 >= B_ * N_) return;
  int half = lane >> 4, l16 = lane & 15;
  v8f acc;
#pragma unroll
  for (int e = 0; e < 8; ++e) acc[e] = 0.0f;
  const float* arow = H + (size_t)(m0 + l16) * NF + 8 * half;
#pragma unroll
  for (int kc = 0; kc < 2; ++kc) {
    v16bf af  = pack_a(arow + kc * 32, 1.0f);
    v16bf bf_ = *(const v16bf*)(W2f + ((size_t)kc * 32 + lane) * 16);
    acc = wmma_bf16(af, bf_, acc);
  }
  if (l16 < 3) {
    float bv = b2[l16];
#pragma unroll
    for (int rr = 0; rr < 8; ++rr) {
      int row = m0 + rr + 8 * half;
      out[(size_t)row * 3 + l16] = states[(size_t)row * 3 + l16] + acc[rr] + bv;
    }
  }
}

// ---------------------------------------------------------------------------
extern "C" void kernel_launch(void* const* d_in, const int* in_sizes, int n_in,
                              void* d_out, int out_size, void* d_ws, size_t ws_size,
                              hipStream_t stream) {
  (void)in_sizes; (void)n_in; (void)out_size; (void)ws_size;
  const float*   states  = (const float*)d_in[0];
  const float*   a_cur   = (const float*)d_in[1];
  const float*   s_delta = (const float*)d_in[2];
  const uint8_t* mask    = (const uint8_t*)d_in[3];
  const uint8_t* tool    = (const uint8_t*)d_in[4];
  // d_in[5] = topk scalar (== 8, compile-time TOPK here)
  const float* pe_w1 = (const float*)d_in[6];  const float* pe_b1 = (const float*)d_in[7];
  const float* pe_w2 = (const float*)d_in[8];  const float* pe_b2 = (const float*)d_in[9];
  const float* re_w1 = (const float*)d_in[10]; const float* re_b1 = (const float*)d_in[11];
  const float* re_w2 = (const float*)d_in[12]; const float* re_b2 = (const float*)d_in[13];
  const float* re_w3 = (const float*)d_in[14]; const float* re_b3 = (const float*)d_in[15];
  const float* rp_w  = (const float*)d_in[16]; const float* rp_b  = (const float*)d_in[17];
  const float* pp_w  = (const float*)d_in[18]; const float* pp_b  = (const float*)d_in[19];
  const float* pr_w1 = (const float*)d_in[20]; const float* pr_b1 = (const float*)d_in[21];
  const float* pr_w2 = (const float*)d_in[22]; const float* pr_b2 = (const float*)d_in[23];
  float* out = (float*)d_out;

  const int BN = B_ * N_;   // 4096
  const int BR = B_ * R_;   // 32768

  char* p = (char*)d_ws;
  auto alloc = [&](size_t bytes) -> char* {
    char* r = p; p += (bytes + 255) & ~(size_t)255; return r;
  };
  int*    idx    = (int*)  alloc((size_t)BN * TOPK * 4);
  float*  valid  = (float*)alloc((size_t)BN * TOPK * 4);
  float*  X32    = (float*)alloc((size_t)BN * 32 * 4);
  float*  Hbuf   = (float*)alloc((size_t)BN * NF * 4);
  float*  effect = (float*)alloc((size_t)BN * NF * 4);
  float*  predh  = (float*)alloc((size_t)BN * NF * 4);
  float*  RI32   = (float*)alloc((size_t)BR * 32 * 4);
  float*  relA   = (float*)alloc((size_t)BR * NF * 4);   // scratch / rel_eff
  float*  relB   = (float*)alloc((size_t)BR * NF * 4);   // scratch
  float*  relC   = (float*)alloc((size_t)BR * NF * 4);   // rel_enc (constant)
  float*  ppcat  = (float*)alloc((size_t)BN * 128 * 4);  // [p_enc | agg]
  __bf16* wpe1 = (__bf16*)alloc((size_t)32 * 64 * 2);
  __bf16* wpe2 = (__bf16*)alloc((size_t)64 * 64 * 2);
  __bf16* wre1 = (__bf16*)alloc((size_t)32 * 64 * 2);
  __bf16* wre2 = (__bf16*)alloc((size_t)64 * 64 * 2);
  __bf16* wre3 = (__bf16*)alloc((size_t)64 * 64 * 2);
  __bf16* wrp  = (__bf16*)alloc((size_t)192 * 64 * 2);
  __bf16* wpp  = (__bf16*)alloc((size_t)128 * 64 * 2);
  __bf16* wpr1 = (__bf16*)alloc((size_t)64 * 64 * 2);
  __bf16* wpr2 = (__bf16*)alloc((size_t)64 * 16 * 2);

  auto swz = [&](const float* src, __bf16* dst, int K0, int N0, int KC, int NT) {
    int tot = KC * NT * 512;
    swz_weight_kernel<<<(tot + 255) / 256, 256, 0, stream>>>(src, dst, K0, N0, KC, NT);
  };
  swz(pe_w1, wpe1, 12, 64, 1, 4);
  swz(pe_w2, wpe2, 64, 64, 2, 4);
  swz(re_w1, wre1, 5, 64, 1, 4);
  swz(re_w2, wre2, 64, 64, 2, 4);
  swz(re_w3, wre3, 64, 64, 2, 4);
  swz(rp_w,  wrp,  192, 64, 6, 4);
  swz(pp_w,  wpp,  128, 64, 4, 4);
  swz(pr_w1, wpr1, 64, 64, 2, 4);
  swz(pr_w2, wpr2, 64, 3, 2, 1);

  // edges
  topk_kernel<<<B_ * (N_ / 256), 256, 0, stream>>>(states, mask, tool, idx, valid);

  // encoder inputs
  build_x_kernel<<<(BN + 255) / 256, 256, 0, stream>>>(s_delta, a_cur, X32);
  build_relin_kernel<<<(BR + 255) / 256, 256, 0, stream>>>(states, a_cur, idx, valid, RI32);

  auto gblk = [](int M) { return M / 256; };  // 8 waves/block * 32 rows/wave

  // particle encoder -> effect (== p_enc)
  gemm_kernel<4, true, false><<<gblk(BN), 256, 0, stream>>>(
      X32, 32, wpe1, pe_b1, Hbuf, 64, nullptr, 0, BN, 32);
  gemm_kernel<4, true, false><<<gblk(BN), 256, 0, stream>>>(
      Hbuf, 64, wpe2, pe_b2, effect, 64, nullptr, 0, BN, 64);
  copy_penc_kernel<<<(BN * NF + 255) / 256, 256, 0, stream>>>(effect, ppcat);

  // relation encoder -> relC (constant across steps)
  gemm_kernel<4, true, false><<<gblk(BR), 256, 0, stream>>>(
      RI32, 32, wre1, re_b1, relA, 64, nullptr, 0, BR, 32);
  gemm_kernel<4, true, false><<<gblk(BR), 256, 0, stream>>>(
      relA, 64, wre2, re_b2, relB, 64, nullptr, 0, BR, 64);
  gemm_kernel<4, true, false><<<gblk(BR), 256, 0, stream>>>(
      relB, 64, wre3, re_b3, relC, 64, nullptr, 0, BR, 64);

  // propagation steps (gather fused into rel GEMM)
  for (int step = 0; step < 3; ++step) {
    rel_gemm_kernel<<<gblk(BR), 256, 0, stream>>>(
        relC, effect, idx, valid, wrp, rp_b, relA /*rel_eff*/);
    aggregate_kernel<<<(BN + 255) / 256, 256, 0, stream>>>(relA, valid, ppcat);
    gemm_kernel<4, true, true><<<gblk(BN), 256, 0, stream>>>(
        ppcat, 128, wpp, pp_b, effect, 64, effect, 64, BN, 128);
  }

  // predictor
  gemm_kernel<4, true, false><<<gblk(BN), 256, 0, stream>>>(
      effect, 64, wpr1, pr_b1, predh, 64, nullptr, 0, BN, 64);
  final_kernel<<<BN / 128, 256, 0, stream>>>(predh, wpr2, pr_b2, states, out);
}
